// CausalShapedAttention_17351667876358
// MI455X (gfx1250) — matrix operand order repaired and checked by reading, not compiled, not run on hardware
//
#include <hip/hip_runtime.h>

typedef __attribute__((ext_vector_type(16))) _Float16 v16h;
typedef __attribute__((ext_vector_type(8)))  _Float16 v8h;
typedef __attribute__((ext_vector_type(8)))  float    v8f;
typedef __attribute__((ext_vector_type(4)))  float    v4f;

union AV { v16h v; v8h h[2]; };

constexpr int Bn = 2, T = 2048, Cc = 1024, H = 16, HD = 64;
constexpr int Mrows = Bn * T;      // 4096
constexpr int Ncols = 2 * Cc;      // 2048

// ---------------------------------------------------------------------------
// Kernel 1: per-channel prefix scan.
//   out[b,t,c] = alpha*x - gamma*(prefix_sum/(t+1));  vh[b,h,t,d] = (f16)x
// ---------------------------------------------------------------------------
__global__ __launch_bounds__(256) void prefix_kernel(
    const float* __restrict__ x, float* __restrict__ out,
    _Float16* __restrict__ vh,
    const float* __restrict__ alpha_p, const float* __restrict__ gamma_p) {
  int idx = blockIdx.x * blockDim.x + threadIdx.x;   // 0 .. Bn*Cc-1
  int b = idx >> 10, c = idx & (Cc - 1);
  int h = c >> 6, d = c & (HD - 1);
  float alpha = *alpha_p, gamma = *gamma_p;
  const float* xp = x  + (size_t)b * T * Cc + c;
  float*       op = out + (size_t)b * T * Cc + c;
  _Float16*    vp = vh + ((size_t)(b * H + h) * T) * HD + d;
  float run = 0.f;
  for (int t = 0; t < T; ++t) {
    float xv = xp[(size_t)t * Cc];
    run += xv;
    op[(size_t)t * Cc] = alpha * xv - gamma * (run / (float)(t + 1));
    vp[(size_t)t * HD] = (_Float16)xv;
  }
}

// ---------------------------------------------------------------------------
// Kernel 2: qk = x @ W^T  (M=4096, N=2048, K=1024), f16 WMMA, fp32 accum.
// Block: 256 threads = 8 waves; tile 64(M) x 128(N); wave = 16M x 64N.
// K staged 64-deep per LDS stage -> 8 WMMAs per barrier pair.
// B fragments batch-loaded before the WMMA chain; next tiles prefetched.
// ---------------------------------------------------------------------------
__global__ __launch_bounds__(256) void qk_gemm(
    const float* __restrict__ x, const float* __restrict__ W,
    _Float16* __restrict__ qh, _Float16* __restrict__ kh) {
  __shared__ __align__(32) _Float16 As[64][80];    // [m][k]  160B rows
  __shared__ __align__(32) _Float16 Bs[128][80];   // [n][k]  160B rows

  const int tid = threadIdx.x, lane = tid & 31, wv = tid >> 5;
  const int wm = wv & 3, wn = wv >> 2;             // 4x2 wave grid
  const int row0 = blockIdx.x * 64;                // M base
  const int col0 = blockIdx.y * 128;               // N base

  // per-thread staging coordinates
  const int ar  = tid >> 2;                        // A row 0..63
  const int akq = (tid & 3) * 16;                  // A k-quad
  const int bnr = tid >> 1;                        // B row 0..127
  const int bkq = (tid & 1) * 32;                  // B k-half

  v8f acc[4] = {};

  for (int k0 = 0; k0 < Cc; k0 += 64) {
    __syncthreads();
    {   // A tile: 64x64 fp32 -> f16; thread loads 16 contiguous floats
      const float* src = x + (size_t)(row0 + ar) * Cc + k0 + akq;
#pragma unroll
      for (int q = 0; q < 4; ++q) {
        v4f f = *(const v4f*)(src + 4 * q);
#pragma unroll
        for (int i = 0; i < 4; ++i) As[ar][akq + 4 * q + i] = (_Float16)f[i];
      }
    }
    {   // B tile: 128x64 fp32 -> f16; thread loads 32 contiguous floats
      const float* src = W + (size_t)(col0 + bnr) * Cc + k0 + bkq;
#pragma unroll
      for (int q = 0; q < 8; ++q) {
        v4f f = *(const v4f*)(src + 4 * q);
#pragma unroll
        for (int i = 0; i < 4; ++i) Bs[bnr][bkq + 4 * q + i] = (_Float16)f[i];
      }
    }
    // prefetch next stage's tiles while this stage computes
    if (k0 + 64 < Cc) {
      __builtin_prefetch(x + (size_t)(row0 + ar) * Cc + k0 + 64 + akq, 0, 0);
      __builtin_prefetch(W + (size_t)(col0 + bnr) * Cc + k0 + 64 + bkq, 0, 0);
    }
    __syncthreads();

    const int am = wm * 16 + (lane & 15);
    const int ak = (lane >> 4) * 16;
#pragma unroll
    for (int ks = 0; ks < 64; ks += 32) {
      // batch-load A + all 4 B fragments, then run the WMMA chain
      AV a;
      a.h[0] = *(const v8h*)&As[am][ks + ak];
      a.h[1] = *(const v8h*)&As[am][ks + ak + 8];
      v16h bf[4];
#pragma unroll
      for (int j = 0; j < 4; ++j) {
        int bn = wn * 64 + j * 16 + (lane & 15);   // output column per lane
        int bk = ks + (lane >> 4) * 16;            // 16 contiguous K
        bf[j] = *(const v16h*)&Bs[bn][bk];
      }
#pragma unroll
      for (int j = 0; j < 4; ++j)
        acc[j] = __builtin_amdgcn_wmma_f32_16x16x32_f16(
            false, a.v, false, bf[j], (short)0, acc[j], false, false);
    }
  }

  // C layout: VGPR r, lanes0-15 -> M=r, lanes16-31 -> M=r+8; N = lane&15
  int mbase = row0 + wm * 16 + ((lane < 16) ? 0 : 8);
#pragma unroll
  for (int j = 0; j < 4; ++j) {
    int o = col0 + wn * 64 + j * 16 + (lane & 15);
    _Float16* dst = (o < Cc) ? qh : kh;
    int c = o & (Cc - 1);
    int h = c >> 6, d = c & (HD - 1);
#pragma unroll
    for (int r = 0; r < 8; ++r) {
      int m = mbase + r;
      int b = m >> 11, t = m & (T - 1);
      dst[(((size_t)(b * H + h)) * T + t) * HD + d] = (_Float16)acc[j][r];
    }
  }
}

// ---------------------------------------------------------------------------
// Kernel 3: causal flash attention, f16 WMMA, fp32 online softmax.
// K tile staged via gfx1250 async LDS copy (ASYNCcnt); V transposed through
// VGPRs; next V tile prefetched. Fragments batch-loaded before WMMA chains.
// ---------------------------------------------------------------------------
__global__ __launch_bounds__(256) void attn_kernel(
    const _Float16* __restrict__ qh, const _Float16* __restrict__ kh,
    const _Float16* __restrict__ vh, float* __restrict__ out,
    const float* __restrict__ beta_p) {
  __shared__ __align__(32) _Float16 Ks[32][80];        // [key][d]   (160B rows)
  __shared__ __align__(32) _Float16 Vs[64][48];        // [d][key]   (96B rows)
  __shared__ __align__(32) _Float16 Ps[8][16][48];     // per-wave P scratch

  const int tid = threadIdx.x, lane = tid & 31, wv = tid >> 5;
  const int bh = blockIdx.y;                 // b*H + h
  const int q0 = blockIdx.x * 128;
  const int qrow0 = q0 + wv * 16;            // wave's 16 q rows

  const _Float16* Q = qh + (size_t)bh * T * HD;
  const _Float16* K = kh + (size_t)bh * T * HD;
  const _Float16* V = vh + (size_t)bh * T * HD;

  // Q fragments (two 16x32 d-windows), kept in registers for all k-blocks
  AV qa[2];
  {
    int am = qrow0 + (lane & 15);
    int ak = (lane >> 4) * 16;
#pragma unroll
    for (int s = 0; s < 2; ++s) {
      const _Float16* src = Q + (size_t)am * HD + s * 32 + ak;
      qa[s].h[0] = *(const v8h*)src;
      qa[s].h[1] = *(const v8h*)(src + 8);
    }
  }

  v8f o_acc[4] = {};
  float m_r[8], l_r[8];
#pragma unroll
  for (int r = 0; r < 8; ++r) { m_r[r] = -3.0e38f; l_r[r] = 0.f; }

  // per-thread tile-staging coordinates (16B of K, 16B of V each)
  const int sr = tid >> 3;                   // key 0..31
  const int sd = (tid & 7) * 8;              // d 0..56

  const int nkb = (q0 + 128) / 32;           // k-blocks incl. last diagonal
  for (int kb = 0; kb < nkb; ++kb) {
    const int kbase = kb * 32;
    __syncthreads();
    {
      // --- K tile: async copy global -> LDS (no VGPR round trip) ---
      unsigned long long kga = (unsigned long long)(K + (size_t)(kbase + sr) * HD + sd);
      unsigned kld = (unsigned)(unsigned long long)(uintptr_t)&Ks[sr][sd];
      asm volatile("global_load_async_to_lds_b128 %0, %1, off"
                   :: "v"(kld), "v"(kga) : "memory");
      // --- V tile: load + transpose into [d][key] ---
      v8h vvv = *(const v8h*)(V + (size_t)(kbase + sr) * HD + sd);
#pragma unroll
      for (int i = 0; i < 8; ++i) Vs[sd + i][sr] = vvv[i];
      // --- prefetch next V tile while this block computes ---
      if (kb + 1 < nkb)
        __builtin_prefetch(V + (size_t)(kbase + 32 + sr) * HD + sd, 0, 0);
      asm volatile("s_wait_asynccnt 0x0" ::: "memory");
    }
    __syncthreads();

    if (kbase <= qrow0 + 15) {               // wave-uniform; no barriers inside
      // ---- scores: S = q k^T (16 x 32 keys), 4 WMMAs, batched frags ----
      v16h kf[4];
#pragma unroll
      for (int nt = 0; nt < 2; ++nt) {
        int key = nt * 16 + (lane & 15);
#pragma unroll
        for (int s = 0; s < 2; ++s) {
          int dk = s * 32 + (lane >> 4) * 16;
          kf[nt * 2 + s] = *(const v16h*)&Ks[key][dk];
        }
      }
      v8f s_acc[2] = {};
#pragma unroll
      for (int nt = 0; nt < 2; ++nt)
#pragma unroll
        for (int s = 0; s < 2; ++s)
          s_acc[nt] = __builtin_amdgcn_wmma_f32_16x16x32_f16(
              false, qa[s].v, false, kf[nt * 2 + s], (short)0, s_acc[nt],
              false, false);
      // ---- scale + causal mask ----
      const int mrow = qrow0 + ((lane < 16) ? 0 : 8);
#pragma unroll
      for (int nt = 0; nt < 2; ++nt) {
        int key = kbase + nt * 16 + (lane & 15);
#pragma unroll
        for (int r = 0; r < 8; ++r) {
          float sv = s_acc[nt][r] * 0.125f;
          s_acc[nt][r] = (key <= mrow + r) ? sv : -3.0e38f;
        }
      }
      // ---- online softmax: row stats via 16-lane xor reductions ----
#pragma unroll
      for (int r = 0; r < 8; ++r) {
        float mx = fmaxf(s_acc[0][r], s_acc[1][r]);
#pragma unroll
        for (int off = 1; off < 16; off <<= 1)
          mx = fmaxf(mx, __shfl_xor(mx, off, 32));
        float mnew = fmaxf(m_r[r], mx);
        float corr = __expf(m_r[r] - mnew);
        m_r[r] = mnew;
        l_r[r] *= corr;
#pragma unroll
        for (int j = 0; j < 4; ++j) o_acc[j][r] *= corr;
        float p0 = __expf(s_acc[0][r] - mnew);
        float p1 = __expf(s_acc[1][r] - mnew);
        s_acc[0][r] = p0; s_acc[1][r] = p1;
        float sm = p0 + p1;
#pragma unroll
        for (int off = 1; off < 16; off <<= 1)
          sm += __shfl_xor(sm, off, 32);
        l_r[r] += sm;
      }
      // ---- C-layout -> A-layout via per-wave LDS scratch ----
#pragma unroll
      for (int nt = 0; nt < 2; ++nt)
#pragma unroll
        for (int r = 0; r < 8; ++r)
          Ps[wv][r + ((lane < 16) ? 0 : 8)][nt * 16 + (lane & 15)] =
              (_Float16)s_acc[nt][r];
      asm volatile("s_wait_dscnt 0x0" ::: "memory");
      __builtin_amdgcn_wave_barrier();
      AV pa;
      {
        int pm = lane & 15;
        int pk = (lane >> 4) * 16;
        pa.h[0] = *(const v8h*)&Ps[wv][pm][pk];
        pa.h[1] = *(const v8h*)&Ps[wv][pm][pk + 8];
      }
      // ---- O += P V  (16 x 64), 4 WMMAs, batched frags ----
      v16h vf[4];
#pragma unroll
      for (int j = 0; j < 4; ++j) {
        int dr = j * 16 + (lane & 15);
        int vk = (lane >> 4) * 16;
        vf[j] = *(const v16h*)&Vs[dr][vk];
      }
#pragma unroll
      for (int j = 0; j < 4; ++j)
        o_acc[j] = __builtin_amdgcn_wmma_f32_16x16x32_f16(
            false, pa.v, false, vf[j], (short)0, o_acc[j], false, false);
    }
  }

  // ---- finalize: out += beta * O / l ----
  const float beta = *beta_p;
  const int b = bh >> 4, h = bh & (H - 1);
  const int mbase = qrow0 + ((lane < 16) ? 0 : 8);
#pragma unroll
  for (int r = 0; r < 8; ++r) {
    float inv = beta / l_r[r];
    int t = mbase + r;
#pragma unroll
    for (int j = 0; j < 4; ++j) {
      int d = j * 16 + (lane & 15);
      size_t oi = ((size_t)b * T + t) * Cc + h * HD + d;
      out[oi] += o_acc[j][r] * inv;
    }
  }
}

// ---------------------------------------------------------------------------
extern "C" void kernel_launch(void* const* d_in, const int* in_sizes, int n_in,
                              void* d_out, int out_size, void* d_ws, size_t ws_size,
                              hipStream_t stream) {
  const float* x     = (const float*)d_in[0];
  const float* W     = (const float*)d_in[1];
  const float* alpha = (const float*)d_in[2];
  const float* beta  = (const float*)d_in[3];
  const float* gamma = (const float*)d_in[4];
  float* out = (float*)d_out;

  const size_t headElems = (size_t)Bn * H * T * HD;   // 4M f16 each
  _Float16* qh = (_Float16*)d_ws;
  _Float16* kh = qh + headElems;
  _Float16* vh = kh + headElems;

  // 1) out = alpha*x - gamma*prefixmean(x); vh = f16(x) in head layout
  prefix_kernel<<<(Bn * Cc) / 256, 256, 0, stream>>>(x, out, vh, alpha, gamma);
  // 2) q,k projection GEMM (f16 WMMA)
  qk_gemm<<<dim3(Mrows / 64, Ncols / 128), 256, 0, stream>>>(x, W, qh, kh);
  // 3) causal flash attention, accumulates beta*att@v into out
  attn_kernel<<<dim3(T / 128, Bn * H), 256, 0, stream>>>(qh, kh, vh, out, beta);
}
